// LSA_55327768707487
// MI455X (gfx1250) — matrix operand order repaired and checked by reading, not compiled, not run on hardware
//
#include <hip/hip_runtime.h>
#include <hip/hip_fp16.h>
#include <stdint.h>

#define B_  32
#define T_  8192
#define AD  128
#define NF  32
#define KW  31

typedef __attribute__((ext_vector_type(16))) _Float16 v16h;
typedef __attribute__((ext_vector_type(8)))  float    v8f;

// Branch-free tanh from hardware transcendentals:
// tanh(x) = 1 - 2/(1+e^{2x});  e^{2x} = exp2(x * 2*log2(e))
// Saturates correctly: exp2->inf => rcp->0 => +1 ; exp2->0 => 1-2*rcp(1) = -1.
__device__ __forceinline__ float fast_tanh(float x) {
  float e = __builtin_amdgcn_exp2f(x * 2.8853900817779268f);
  return 1.0f - 2.0f * __builtin_amdgcn_rcpf(e + 1.0f);
}

__device__ __forceinline__ float fast_exp(float x) {
  return __builtin_amdgcn_exp2f(x * 1.4426950408889634f);
}

// ---------------- Kernel 1: pq[b][a] = dot(query[b,:], W_w[a,:]) + W_b[a] ----------------
__global__ __launch_bounds__(AD)
void pq_kernel(const float* __restrict__ query,
               const float* __restrict__ Ww,
               const float* __restrict__ Wb,
               float* __restrict__ pq) {
  int b = blockIdx.x, a = threadIdx.x;
  __shared__ float qs[AD];
  qs[a] = query[b * AD + a];
  __syncthreads();
  const float* wr = Ww + a * AD;
  float s = Wb[a];
#pragma unroll 8
  for (int k = 0; k < AD; ++k) s += qs[k] * wr[k];
  pq[b * AD + a] = s;
}

// ---------------- Kernel 2: conv (pk-f16) -> WMMA GEMM -> tanh/dot epilogue -> u ----------------
__global__ __launch_bounds__(256)
void attn_kernel(const float* __restrict__ enc,
                 const float* __restrict__ cum,
                 const int*   __restrict__ chars,
                 const float* __restrict__ conv_w,
                 const float* __restrict__ conv_b,
                 const float* __restrict__ Lw,
                 const float* __restrict__ vw,
                 const float* __restrict__ pq,
                 float* __restrict__ u) {
  const int b   = blockIdx.y;
  const int bt0 = blockIdx.x * 128;          // block covers 128 t's
  const int tid = threadIdx.x;
  const int lane = tid & 31, wave = tid >> 5;

  __shared__ __align__(16) __half    s_cumh[160];      // cum[bt0-15 .. bt0+142] in f16
  __shared__ __align__(16) __half    s_wT[KW * NF];    // wT[k][f] f16
  __shared__ __align__(16) __half    s_cb[NF];         // conv bias f16
  __shared__ __align__(16) uint32_t  s_Bu[2048];       // B-operand image: 8 chunks x 32 lanes x 8 dwords
  __shared__ __align__(16) float     s_pq[AD];
  __shared__ __align__(16) float     s_v[AD];

  // ---- stage block-shared data ----
  for (int i = tid; i < 158; i += 256) {
    int t = bt0 - 15 + i;
    float v = (t >= 0 && t < T_) ? cum[b * T_ + t] : 0.f;
    s_cumh[i] = __float2half(v);
  }
  for (int i = tid; i < KW * NF; i += 256) {
    int k = i >> 5, f = i & 31;
    s_wT[i] = __float2half(conv_w[f * KW + k]);
  }
  if (tid < NF) s_cb[tid] = __float2half(conv_b[tid]);
  if (tid < AD) { s_pq[tid] = pq[b * AD + tid]; s_v[tid] = vw[tid]; }
  // B-matrix (K=filter 32, N=attn 16 per chunk): lanes 0-15 hold K={0..7,16..23} (2/dword),
  // lanes 16-31 hold K={8..15,24..31}, column N = lane%16, value = L_w[a][f].
  for (int i = tid; i < 2048; i += 256) {
    int c = i >> 8;            // chunk
    int r = (i >> 3) & 31;     // lane slot
    int q = i & 7;             // dword within lane
    int a = c * 16 + (r & 15);
    int f = 2 * q + ((q >= 4) ? 8 : 0) + ((r < 16) ? 0 : 8);
    __half2 h2 = __halves2half2(__float2half(Lw[a * NF + f]),
                                __float2half(Lw[a * NF + f + 1]));
    s_Bu[i] = *reinterpret_cast<uint32_t*>(&h2);
  }
  __syncthreads();

  const int t0    = bt0 + wave * 16;
  const int mrow  = lane & 15;
  const int fbase = (lane < 16) ? 0 : 8;

  // ---- conv accumulated in packed f16, directly in A-operand layout ----
  // A (16-bit, 16x32): lane l<16 holds row m=l, K = {0..7,16..23}; l>=16 holds m=l-16, K={8..15,24..31}.
  __half2 accp[8];
#pragma unroll
  for (int j = 0; j < 8; ++j) {
    int flo = fbase + 2 * j + ((j >= 4) ? 8 : 0);
    accp[j] = *reinterpret_cast<const __half2*>(&s_cb[flo]);
  }
#pragma unroll 1
  for (int k = 0; k < KW; ++k) {
    __half2 c2 = __half2half2(s_cumh[wave * 16 + mrow + k]);
#pragma unroll
    for (int j = 0; j < 8; ++j) {
      int flo = fbase + 2 * j + ((j >= 4) ? 8 : 0);
      accp[j] = __hfma2(*reinterpret_cast<const __half2*>(&s_wT[k * NF + flo]), c2, accp[j]);
    }
  }
  v16h A;
  uint32_t* au = reinterpret_cast<uint32_t*>(&A);
#pragma unroll
  for (int j = 0; j < 8; ++j) au[j] = *reinterpret_cast<uint32_t*>(&accp[j]);

  float uacc[8];
#pragma unroll
  for (int j = 0; j < 8; ++j) uacc[j] = 0.f;

  const int trow = t0 + ((lane < 16) ? 0 : 8);           // D rows: VGPR j -> t0+j (lo half) / t0+8+j (hi half)
  const float* encb = enc + ((long)b * T_ + trow) * AD;

  // ---- 8 WMMAs cover the 128 attention dims ----
#pragma unroll 2
  for (int c = 0; c < 8; ++c) {
    v16h Bv;
    uint32_t* bu = reinterpret_cast<uint32_t*>(&Bv);
#pragma unroll
    for (int q = 0; q < 8; ++q) bu[q] = s_Bu[c * 256 + lane * 8 + q];
    v8f C = {};
    v8f D = __builtin_amdgcn_wmma_f32_16x16x32_f16(false, A, false, Bv,
                                                   (short)0, C, false, false);
    int a = c * 16 + mrow;
    float pqa = s_pq[a];
    float va  = s_v[a];
    const float* ep = encb + a;
#pragma unroll
    for (int j = 0; j < 8; ++j) {
      float x = D[j] + pqa + ep[j * AD];
      uacc[j] += va * fast_tanh(x);
    }
  }

  // ---- reduce over attn dim: butterfly within each 16-lane half ----
#pragma unroll
  for (int j = 0; j < 8; ++j) {
    float s = uacc[j];
    s += __shfl_xor(s, 1, 32);
    s += __shfl_xor(s, 2, 32);
    s += __shfl_xor(s, 4, 32);
    s += __shfl_xor(s, 8, 32);
    if (mrow == 0) {
      int t = t0 + j + ((lane >> 4) << 3);
      u[b * T_ + t] = (chars[b * T_ + t] != 0) ? s : 0.f;
    }
  }
}

// ---------------- Kernel 3: in-place per-batch softmax over T ----------------
__global__ __launch_bounds__(256)
void softmax_kernel(float* __restrict__ uo) {
  int b = blockIdx.x, tid = threadIdx.x;
  __shared__ float red[256];
  float* row = uo + (long)b * T_;

  float m = -1e30f;
  for (int t = tid; t < T_; t += 256) m = fmaxf(m, row[t]);
  red[tid] = m; __syncthreads();
  for (int s = 128; s > 0; s >>= 1) {
    if (tid < s) red[tid] = fmaxf(red[tid], red[tid + s]);
    __syncthreads();
  }
  float mx = red[0]; __syncthreads();

  float sum = 0.f;
  for (int t = tid; t < T_; t += 256) {
    float e = fast_exp(row[t] - mx);
    row[t] = e;                 // each t owned by exactly one thread
    sum += e;
  }
  red[tid] = sum; __syncthreads();
  for (int s = 128; s > 0; s >>= 1) {
    if (tid < s) red[tid] += red[tid + s];
    __syncthreads();
  }
  float inv = 1.f / red[0];

  for (int t = tid; t < T_; t += 256) row[t] *= inv;
}

// ---------------- launcher ----------------
extern "C" void kernel_launch(void* const* d_in, const int* in_sizes, int n_in,
                              void* d_out, int out_size, void* d_ws, size_t ws_size,
                              hipStream_t stream) {
  const float* enc    = (const float*)d_in[0];
  const float* query  = (const float*)d_in[1];
  const float* cum    = (const float*)d_in[2];
  const int*   chars  = (const int*)d_in[3];
  /* d_in[4] = t (unused) */
  const float* conv_w = (const float*)d_in[5];
  const float* conv_b = (const float*)d_in[6];
  const float* Lw     = (const float*)d_in[7];
  const float* Ww     = (const float*)d_in[8];
  const float* Wb     = (const float*)d_in[9];
  const float* vw     = (const float*)d_in[10];
  (void)in_sizes; (void)n_in; (void)out_size; (void)ws_size;

  float* out = (float*)d_out;            // u then scores, in place (exactly B*T floats)
  float* pq  = (float*)d_ws;             // 32*128 floats of scratch

  pq_kernel<<<B_, AD, 0, stream>>>(query, Ww, Wb, pq);
  dim3 grid(T_ / 128, B_);
  attn_kernel<<<grid, 256, 0, stream>>>(enc, cum, chars, conv_w, conv_b, Lw, vw, pq, out);
  softmax_kernel<<<B_, 256, 0, stream>>>(out);
}
